// SceneGraphEmbedding_56822417326514
// MI455X (gfx1250) — compile-verified, matrix operand order checked by reading, not run on hardware
//
#include <hip/hip_runtime.h>
#include <cstdint>

typedef __attribute__((ext_vector_type(16))) __bf16       v16bf;
typedef __attribute__((ext_vector_type(8)))  float        v8f;
typedef __attribute__((ext_vector_type(8)))  unsigned int v8u;

#define NUM_GRAPHS 64

__device__ inline unsigned int pk_bf16(float a, float b) {
  __bf16 x = (__bf16)a, y = (__bf16)b;
  return (unsigned int)__builtin_bit_cast(unsigned short, x) |
         ((unsigned int)__builtin_bit_cast(unsigned short, y) << 16);
}

// ---------------- utility kernels ----------------
__global__ void fill_f32(float* p, float v, size_t n) {
  size_t i = (size_t)blockIdx.x * blockDim.x + threadIdx.x;
  if (i < n) p[i] = v;
}

// Pack W[K x M] f32 -> Wpk[Kpairs x M] u32, each u32 = {bf16(W[2p]), bf16(W[2p+1])},
// zero-padded beyond K: this makes the GEMM's A-side K-padding mathematically inert.
__global__ void cvt_pack_w(const float* __restrict__ W, unsigned int* __restrict__ Wpk,
                           int K, int M, int Kpairs) {
  int i = blockIdx.x * blockDim.x + threadIdx.x;
  if (i >= Kpairs * M) return;
  int p = i / M, col = i % M;
  int k0 = 2 * p, k1 = 2 * p + 1;
  float a = (k0 < K) ? W[(size_t)k0 * M + col] : 0.f;
  float b = (k1 < K) ? W[(size_t)k1 * M + col] : 0.f;
  Wpk[i] = pk_bf16(a, b);
}

// h0 layout: [cont(16) | s(32, filled by GEMM) | zero pad(16)]  => 64 cols
__global__ void build_h0(const float* __restrict__ x, float* __restrict__ h0, int n) {
  int i = blockIdx.x * blockDim.x + threadIdx.x;
  if (i >= n * 64) return;
  int node = i >> 6, j = i & 63;
  if (j < 16)       h0[i] = x[node * 40 + 24 + j];
  else if (j >= 48) h0[i] = 0.f;
}

// ---------------- WMMA GEMM: C[n x M] = act(A_f32[n x lda] @ W_bf16[K x M] + bias) ----------------
// CONTRACT: ceil(K/32)*32 <= lda (full 32-col stripes always in-bounds & finite);
//           weights zero-padded to pair boundary, so A garbage in [K, kpad) contributes 0.
// block = 256 threads (8 waves); tile = 128 rows x 16 cols; unconditional vectorized staging.
__global__ __launch_bounds__(256)
void wmma_gemm(const float* __restrict__ A, int lda,
               const unsigned int* __restrict__ Wpk, const float* __restrict__ bias,
               float* __restrict__ C, int ldc, int coff,
               int nrows, int K, int M, int act) {
  __shared__ unsigned int sA[128][17];   // [row][pair], padded pitch (bank-conflict free)
  __shared__ unsigned int sB[128][16];   // [pair][col]

  const int t    = threadIdx.x;
  const int lane = t & 31;
  const int wave = t >> 5;
  const int rowBlk = blockIdx.x * 128;
  const int colBlk = blockIdx.y * 16;
  const int m = lane & 15;               // A row in tile / B,C column
  const int g = lane >> 4;               // lane group

  const int ksteps = (K + 31) >> 5;
  const int Kp = ksteps << 4;            // total pairs (multiple of 16)

  // stage whole B panel once (coalesced u32 copies, already padded)
  for (int idx = t; idx < Kp * 16; idx += 256) {
    const int p = idx >> 4, nn = idx & 15;
    sB[p][nn] = Wpk[(size_t)p * M + colBlk + nn];
  }

  // A staging address: 2 threads per row, 16 cols each; clamp tail rows to a valid row
  const int r  = t >> 1;
  const int cb = (t & 1) << 4;           // 0 or 16
  int grow = rowBlk + r;
  if (grow >= nrows) grow = nrows - 1;
  const float* arow = A + (size_t)grow * lda + cb;
  unsigned int* sdst = &sA[r][cb >> 1];

  v8f acc = {};
  for (int ks = 0; ks < ksteps; ++ks) {
    // ---- stage A tile: 128 rows x 32 cols -> bf16 pairs (always full-width vector path)
    {
      const float4* ap = (const float4*)(arow + (ks << 5));
#pragma unroll
      for (int q = 0; q < 4; ++q) {
        float4 f = ap[q];
        sdst[2 * q]     = pk_bf16(f.x, f.y);
        sdst[2 * q + 1] = pk_bf16(f.z, f.w);
      }
    }
    __syncthreads();

    // ---- fragments (ISA 7.12.2 pair layout), then WMMA
    v8u au, bu;
    const int k0p = ks << 4;
#pragma unroll
    for (int v = 0; v < 8; ++v) {
      const int pidx = (v < 4) ? (v + 4 * g) : (8 + (v - 4) + 4 * g);
      au[v] = sA[wave * 16 + m][pidx];
      bu[v] = sB[k0p + pidx][m];
    }
    v16bf af = __builtin_bit_cast(v16bf, au);
    v16bf bf = __builtin_bit_cast(v16bf, bu);
    acc = __builtin_amdgcn_wmma_f32_16x16x32_bf16(false, af, false, bf,
                                                  (short)0, acc, false, false);
    __syncthreads();
  }

  // ---- epilogue: VGPR r holds row (r + 8*g), col = lane&15
  const int col = colBlk + m;
  const float bb = bias ? bias[col] : 0.f;
#pragma unroll
  for (int rr = 0; rr < 8; ++rr) {
    const int row = rowBlk + wave * 16 + rr + 8 * g;
    if (row < nrows) {
      float v = acc[rr] + bb;
      if (act == 1)      v = v > 0.f ? v : 0.f;               // relu
      else if (act == 2) v = v > 0.f ? v : (__expf(v) - 1.f); // elu
      C[(size_t)row * ldc + coff + col] = v;
    }
  }
}

// ---------------- GATv2 edge phase ----------------
__device__ inline void atomicMaxF32(float* addr, float val) {
  if (val >= 0.f) atomicMax((int*)addr, __float_as_int(val));
  else            atomicMin((unsigned int*)addr, __float_as_uint(val));
}

__global__ void edge_logits(const float* __restrict__ xl, const float* __restrict__ xr,
                            const float* __restrict__ att, const int* __restrict__ ei,
                            float* __restrict__ logit, float* __restrict__ maxv,
                            int E, int Etot, int H, int C) {
  const long long tid = (long long)blockIdx.x * blockDim.x + threadIdx.x;
  if (tid >= (long long)Etot * H) return;
  const int e = (int)(tid / H), h = (int)(tid % H);
  int src, dst;
  if (e < E) { src = ei[e]; dst = ei[E + e]; } else { src = dst = e - E; }
  const int D = H * C;
  const float4* pl = (const float4*)(xl + (size_t)src * D + h * C);
  const float4* pr = (const float4*)(xr + (size_t)dst * D + h * C);
  const float4* pa = (const float4*)(att + h * C);
  float acc = 0.f;
  for (int c = 0; c < (C >> 2); ++c) {
    float4 a = pl[c], b = pr[c], w = pa[c];
    float v0 = a.x + b.x, v1 = a.y + b.y, v2 = a.z + b.z, v3 = a.w + b.w;
    v0 = v0 > 0.f ? v0 : 0.2f * v0;
    v1 = v1 > 0.f ? v1 : 0.2f * v1;
    v2 = v2 > 0.f ? v2 : 0.2f * v2;
    v3 = v3 > 0.f ? v3 : 0.2f * v3;
    acc += w.x * v0 + w.y * v1 + w.z * v2 + w.w * v3;
  }
  logit[tid] = acc;
  atomicMaxF32(&maxv[(long long)dst * H + h], acc);
}

__global__ void edge_exp(const int* __restrict__ ei, float* __restrict__ logit,
                         const float* __restrict__ maxv, float* __restrict__ denom,
                         int E, int Etot, int H) {
  const long long tid = (long long)blockIdx.x * blockDim.x + threadIdx.x;
  if (tid >= (long long)Etot * H) return;
  const int e = (int)(tid / H);
  const int h = (int)(tid % H);
  int dst = (e < E) ? ei[E + e] : (e - E);
  float ee = __expf(logit[tid] - maxv[(long long)dst * H + h]);
  logit[tid] = ee;
  atomicAdd(&denom[(long long)dst * H + h], ee);
}

// one block per edge, one thread per channel (blockDim == H*C)
__global__ void edge_scatter(const float* __restrict__ xl, const int* __restrict__ ei,
                             const float* __restrict__ logit, const float* __restrict__ denom,
                             float* __restrict__ outacc, int E, int H, int C) {
  const int e = blockIdx.x, c = threadIdx.x, h = c / C;
  int src, dst;
  if (e < E) { src = ei[e]; dst = ei[E + e]; } else { src = dst = e - E; }
  const float alpha = logit[(long long)e * H + h] / (denom[(long long)dst * H + h] + 1e-16f);
  const int D = H * C;
  atomicAdd(&outacc[(long long)dst * D + c], alpha * xl[(long long)src * D + c]);
}

__global__ void bias_elu(float* __restrict__ g, const float* __restrict__ bias, int D, size_t n) {
  size_t i = (size_t)blockIdx.x * blockDim.x + threadIdx.x;
  if (i >= n) return;
  float v = g[i] + bias[i % D];
  g[i] = v > 0.f ? v : (__expf(v) - 1.f);
}

// ---------------- pooling ----------------
__global__ void pool_accum(const float* __restrict__ g, const int* __restrict__ batch,
                           float* __restrict__ sums, float* __restrict__ cnts, int n, int D) {
  const long long i = (long long)blockIdx.x * blockDim.x + threadIdx.x;
  if (i >= (long long)n * D) return;
  const int node = (int)(i / D), c = (int)(i % D);
  const int b = batch[node];
  atomicAdd(&sums[b * D + c], g[i]);
  if (c == 0) atomicAdd(&cnts[b], 1.f);
}

__global__ void pool_div(const float* __restrict__ sums, const float* __restrict__ cnts,
                         float* __restrict__ pooled, int D) {
  int i = blockIdx.x * blockDim.x + threadIdx.x;
  if (i >= NUM_GRAPHS * D) return;
  pooled[i] = sums[i] / fmaxf(cnts[i / D], 1.f);
}

// ---------------- launch ----------------
static inline int kpairs_of(int K) { return ((K + 31) / 32) * 16; }

extern "C" void kernel_launch(void* const* d_in, const int* in_sizes, int n_in,
                              void* d_out, int out_size, void* d_ws, size_t ws_size,
                              hipStream_t stream) {
  const float* x       = (const float*)d_in[0];
  const int*   ei      = (const int*)d_in[1];
  const int*   batch   = (const int*)d_in[2];
  const float* cat_W   = (const float*)d_in[3];
  const float* cat_b   = (const float*)d_in[4];
  const float* init_W  = (const float*)d_in[5];
  const float* init_b  = (const float*)d_in[6];
  const float* c1_Wl   = (const float*)d_in[7];
  const float* c1_bl   = (const float*)d_in[8];
  const float* c1_Wr   = (const float*)d_in[9];
  const float* c1_br   = (const float*)d_in[10];
  const float* c1_att  = (const float*)d_in[11];
  const float* c1_bias = (const float*)d_in[12];
  const float* c2_Wl   = (const float*)d_in[13];
  const float* c2_bl   = (const float*)d_in[14];
  const float* c2_Wr   = (const float*)d_in[15];
  const float* c2_br   = (const float*)d_in[16];
  const float* c2_att  = (const float*)d_in[17];
  const float* c2_bias = (const float*)d_in[18];
  const float* out_W   = (const float*)d_in[19];
  const float* out_b   = (const float*)d_in[20];

  const int n    = in_sizes[0] / 40;
  const int E    = in_sizes[1] / 2;
  const int Etot = E + n;

  char* base = (char*)d_ws;
  size_t off = 0;
  auto alloc = [&](size_t bytes) -> size_t {
    size_t o = off; off += (bytes + 255) & ~(size_t)255; return o;
  };
  // packed bf16-pair weights (u32 per pair)
  const int kpCat = kpairs_of(24), kpInit = kpairs_of(48), kpC1 = kpairs_of(64),
            kpC2 = kpairs_of(256), kpOut = kpairs_of(64);
  const size_t o_catW = alloc((size_t)kpCat * 32 * 4),
               o_initW = alloc((size_t)kpInit * 64 * 4),
               o_c1Wl = alloc((size_t)kpC1 * 256 * 4), o_c1Wr = alloc((size_t)kpC1 * 256 * 4),
               o_c2Wl = alloc((size_t)kpC2 * 64 * 4),  o_c2Wr = alloc((size_t)kpC2 * 64 * 4),
               o_outW = alloc((size_t)kpOut * 64 * 4);
  // phase-1 f32 buffers
  const size_t o_h0  = alloc((size_t)n * 64 * 4);
  const size_t o_h   = alloc((size_t)n * 64 * 4);
  const size_t o_xl1 = alloc((size_t)n * 256 * 4);
  const size_t o_xr1 = alloc((size_t)n * 256 * 4);
  const size_t o_g1  = alloc((size_t)n * 256 * 4);
  const size_t o_lg1 = alloc((size_t)Etot * 4 * 4);
  const size_t o_mx1 = alloc((size_t)n * 4 * 4);
  const size_t o_dn1 = alloc((size_t)n * 4 * 4);
  // phase-2 aliases onto dead phase-1 buffers
  const size_t o_xl2 = o_h0, o_xr2 = o_h, o_g2 = o_lg1, o_lg2 = o_xl1;
  const size_t o_mx2 = o_mx1, o_dn2 = o_dn1;
  const size_t o_sum = o_xr1;
  const size_t o_cnt = o_xr1 + 64 * 64 * 4;
  const size_t o_pool = o_xr1 + 64 * 64 * 4 + 512;

  unsigned int* catW_pk  = (unsigned int*)(base + o_catW);
  unsigned int* initW_pk = (unsigned int*)(base + o_initW);
  unsigned int* c1Wl_pk  = (unsigned int*)(base + o_c1Wl);
  unsigned int* c1Wr_pk  = (unsigned int*)(base + o_c1Wr);
  unsigned int* c2Wl_pk  = (unsigned int*)(base + o_c2Wl);
  unsigned int* c2Wr_pk  = (unsigned int*)(base + o_c2Wr);
  unsigned int* outW_pk  = (unsigned int*)(base + o_outW);
  float* h0  = (float*)(base + o_h0);
  float* h   = (float*)(base + o_h);
  float* xl1 = (float*)(base + o_xl1);
  float* xr1 = (float*)(base + o_xr1);
  float* g1  = (float*)(base + o_g1);
  float* lg1 = (float*)(base + o_lg1);
  float* mx1 = (float*)(base + o_mx1);
  float* dn1 = (float*)(base + o_dn1);
  float* xl2 = (float*)(base + o_xl2);
  float* xr2 = (float*)(base + o_xr2);
  float* g2  = (float*)(base + o_g2);
  float* lg2 = (float*)(base + o_lg2);
  float* mx2 = (float*)(base + o_mx2);
  float* dn2 = (float*)(base + o_dn2);
  float* sums = (float*)(base + o_sum);
  float* cnts = (float*)(base + o_cnt);
  float* pooled = (float*)(base + o_pool);

  // 1) pack weights into WMMA pair layout (zero K-padded)
  auto packLaunch = [&](const float* W, unsigned int* Wpk, int K, int M) {
    int kp = kpairs_of(K);
    int tot = kp * M;
    cvt_pack_w<<<(tot + 255) / 256, 256, 0, stream>>>(W, Wpk, K, M, kp);
  };
  packLaunch(cat_W, catW_pk, 24, 32);
  packLaunch(init_W, initW_pk, 48, 64);
  packLaunch(c1_Wl, c1Wl_pk, 64, 256);
  packLaunch(c1_Wr, c1Wr_pk, 64, 256);
  packLaunch(c2_Wl, c2Wl_pk, 256, 64);
  packLaunch(c2_Wr, c2Wr_pk, 256, 64);
  packLaunch(out_W, outW_pk, 64, 64);

  // 2) node embedding
  build_h0<<<((size_t)n * 64 + 255) / 256, 256, 0, stream>>>(x, h0, n);
  const unsigned gx = (unsigned)((n + 127) / 128);
  wmma_gemm<<<dim3(gx, 2), 256, 0, stream>>>(x, 40, catW_pk, cat_b, h0, 64, 16, n, 24, 32, 1);
  wmma_gemm<<<dim3(gx, 4), 256, 0, stream>>>(h0, 64, initW_pk, init_b, h, 64, 0, n, 48, 64, 1);

  // 3) conv1 (H=4, C=64)
  fill_f32<<<((size_t)n * 4 + 255) / 256, 256, 0, stream>>>(mx1, -1e30f, (size_t)n * 4);
  fill_f32<<<((size_t)n * 4 + 255) / 256, 256, 0, stream>>>(dn1, 0.f, (size_t)n * 4);
  fill_f32<<<((size_t)n * 256 + 255) / 256, 256, 0, stream>>>(g1, 0.f, (size_t)n * 256);
  wmma_gemm<<<dim3(gx, 16), 256, 0, stream>>>(h, 64, c1Wl_pk, c1_bl, xl1, 256, 0, n, 64, 256, 0);
  wmma_gemm<<<dim3(gx, 16), 256, 0, stream>>>(h, 64, c1Wr_pk, c1_br, xr1, 256, 0, n, 64, 256, 0);
  const size_t t1 = (size_t)Etot * 4;
  edge_logits<<<(t1 + 255) / 256, 256, 0, stream>>>(xl1, xr1, c1_att, ei, lg1, mx1, E, Etot, 4, 64);
  edge_exp<<<(t1 + 255) / 256, 256, 0, stream>>>(ei, lg1, mx1, dn1, E, Etot, 4);
  edge_scatter<<<Etot, 256, 0, stream>>>(xl1, ei, lg1, dn1, g1, E, 4, 64);
  bias_elu<<<((size_t)n * 256 + 255) / 256, 256, 0, stream>>>(g1, c1_bias, 256, (size_t)n * 256);

  // 4) conv2 (H=1, C=64)
  fill_f32<<<((size_t)n + 255) / 256, 256, 0, stream>>>(mx2, -1e30f, (size_t)n);
  fill_f32<<<((size_t)n + 255) / 256, 256, 0, stream>>>(dn2, 0.f, (size_t)n);
  fill_f32<<<((size_t)n * 64 + 255) / 256, 256, 0, stream>>>(g2, 0.f, (size_t)n * 64);
  wmma_gemm<<<dim3(gx, 4), 256, 0, stream>>>(g1, 256, c2Wl_pk, c2_bl, xl2, 64, 0, n, 256, 64, 0);
  wmma_gemm<<<dim3(gx, 4), 256, 0, stream>>>(g1, 256, c2Wr_pk, c2_br, xr2, 64, 0, n, 256, 64, 0);
  edge_logits<<<((size_t)Etot + 255) / 256, 256, 0, stream>>>(xl2, xr2, c2_att, ei, lg2, mx2, E, Etot, 1, 64);
  edge_exp<<<((size_t)Etot + 255) / 256, 256, 0, stream>>>(ei, lg2, mx2, dn2, E, Etot, 1);
  edge_scatter<<<Etot, 64, 0, stream>>>(xl2, ei, lg2, dn2, g2, E, 1, 64);
  bias_elu<<<((size_t)n * 64 + 255) / 256, 256, 0, stream>>>(g2, c2_bias, 64, (size_t)n * 64);

  // 5) mean pool + output GEMM
  fill_f32<<<(64 * 64 + 255) / 256, 256, 0, stream>>>(sums, 0.f, 64 * 64);
  fill_f32<<<1, 64, 0, stream>>>(cnts, 0.f, 64);
  pool_accum<<<((size_t)n * 64 + 255) / 256, 256, 0, stream>>>(g2, batch, sums, cnts, n, 64);
  pool_div<<<(64 * 64 + 255) / 256, 256, 0, stream>>>(sums, cnts, pooled, 64);
  wmma_gemm<<<dim3(1, 4), 256, 0, stream>>>(pooled, 64, outW_pk, out_b,
                                            (float*)d_out, 64, 0, NUM_GRAPHS, 64, 64, 0);
}
